// MultiHeadLatentAttention_44014824849510
// MI455X (gfx1250) — matrix operand order repaired and checked by reading, not compiled, not run on hardware
//
#include <hip/hip_runtime.h>

typedef __attribute__((ext_vector_type(16))) __bf16 v16bf;
typedef __attribute__((ext_vector_type(8)))  float  v8f;
typedef __attribute__((ext_vector_type(4)))  int    v4i;

union FragU { v16bf v; uint4 q[2]; };

__device__ __forceinline__ unsigned short f2bf(float f) {
  union { float f; unsigned u; } c; c.f = f;
  unsigned u = c.u;
  return (unsigned short)((u + 0x7FFFu + ((u >> 16) & 1u)) >> 16);
}
__device__ __forceinline__ float bf2f(unsigned short h) {
  union { unsigned u; float f; } c; c.u = ((unsigned)h) << 16;
  return c.f;
}

// ---- CDNA5 async global->LDS path (guarded; falls back to sync staging) ----
#ifndef __has_builtin
#define __has_builtin(x) 0
#endif

#if __has_builtin(__builtin_amdgcn_global_load_async_to_lds_b128)
#define USE_ASYNC_LDS 1
#else
#define USE_ASYNC_LDS 0
#endif

#if USE_ASYNC_LDS
// prototype (from compiler diagnostic): (v4i __device__* src, v4i __shared__* dst, imm, imm)
#define ASYNC_CP_B128(src, dst) \
  __builtin_amdgcn_global_load_async_to_lds_b128((v4i*)(src), (v4i*)(dst), 0, 0)
#endif

__device__ __forceinline__ void wait_async0() {
#if USE_ASYNC_LDS
#if __has_builtin(__builtin_amdgcn_s_wait_asynccnt)
  __builtin_amdgcn_s_wait_asynccnt(0);
#else
  asm volatile("s_wait_asynccnt 0x0" ::: "memory");
#endif
#endif
}

// ---------------------------------------------------------------------------
// fp32 -> bf16 (RNE) bulk convert: one-time pass over x and all weights so
// every later tile move is a bf16 b128 async DMA.
// ---------------------------------------------------------------------------
__global__ void __launch_bounds__(256)
cvt_f32_bf16(const float* __restrict__ src, unsigned short* __restrict__ dst, int n8)
{
  int i = blockIdx.x * blockDim.x + threadIdx.x;
  if (i >= n8) return;
  const float4* s = (const float4*)src;
  float4 a = s[2 * i], b = s[2 * i + 1];
  uint4 p;
  p.x = (unsigned)f2bf(a.x) | ((unsigned)f2bf(a.y) << 16);
  p.y = (unsigned)f2bf(a.z) | ((unsigned)f2bf(a.w) << 16);
  p.z = (unsigned)f2bf(b.x) | ((unsigned)f2bf(b.y) << 16);
  p.w = (unsigned)f2bf(b.z) | ((unsigned)f2bf(b.w) << 16);
  ((uint4*)dst)[i] = p;
}

// ---------------------------------------------------------------------------
// GEMM: C[M,N] = A[M,K] @ W[N,K]^T   (bf16 in, fp32 accumulate via WMMA)
// Block tile 128x128, K-step 64, 8 waves each owning a 32x64 tile.
// Double-buffered LDS staging: async DMA of tile k+1 overlaps WMMAs on tile k.
// LDS rows padded to 72 elems (144B = 9*16B) so b128 frag loads stay aligned.
// Dynamic LDS: 2 buffers * (A 128x72 + B 128x72) * 2B = 73728 bytes.
// ---------------------------------------------------------------------------
template<bool OUT_F32>
__global__ void __launch_bounds__(256)
gemm_bf16_wmma(const unsigned short* __restrict__ A, const unsigned short* __restrict__ W,
               void* __restrict__ Cp, int M, int N, int K)
{
  extern __shared__ unsigned short smem[];
  // layout: As[2][128*72] then Bs[2][128*72]
  const int t    = threadIdx.x;
  const int lane = t & 31;
  const int wave = t >> 5;
  const int g    = lane >> 4;
  const int ln   = lane & 15;
  const int wm   = wave >> 1;
  const int wn   = wave & 1;
  const int m0   = blockIdx.y * 128;
  const int n0   = blockIdx.x * 128;

  auto stage = [&](int k0, int buf) {
    unsigned short* Ad = smem + buf * 9216;
    unsigned short* Bd = smem + 2 * 9216 + buf * 9216;
#if USE_ASYNC_LDS
#pragma unroll
    for (int i = 0; i < 4; i++) {
      int c = i * 256 + t;
      int row = c >> 3, c8 = (c & 7) * 8;
      ASYNC_CP_B128(A + (size_t)(m0 + row) * K + k0 + c8, &Ad[row * 72 + c8]);
    }
#pragma unroll
    for (int i = 0; i < 4; i++) {
      int c = i * 256 + t;
      int row = c >> 3, c8 = (c & 7) * 8;
      ASYNC_CP_B128(W + (size_t)(n0 + row) * K + k0 + c8, &Bd[row * 72 + c8]);
    }
#else
#pragma unroll
    for (int i = 0; i < 4; i++) {
      int c = i * 256 + t;
      int row = c >> 3, c8 = (c & 7) * 8;
      uint4 d = *(const uint4*)(A + (size_t)(m0 + row) * K + k0 + c8);
      *(uint4*)&Ad[row * 72 + c8] = d;
    }
#pragma unroll
    for (int i = 0; i < 4; i++) {
      int c = i * 256 + t;
      int row = c >> 3, c8 = (c & 7) * 8;
      uint4 d = *(const uint4*)(W + (size_t)(n0 + row) * K + k0 + c8);
      *(uint4*)&Bd[row * 72 + c8] = d;
    }
#endif
  };

  const v8f vzero = {};
  v8f acc[2][4];
#pragma unroll
  for (int i = 0; i < 2; i++)
#pragma unroll
    for (int j = 0; j < 4; j++) acc[i][j] = vzero;

  stage(0, 0);

  int buf = 0;
  for (int k0 = 0; k0 < K; k0 += 64, buf ^= 1) {
    wait_async0();        // this wave's DMA into `buf` is done
    __syncthreads();      // everyone's DMA done; prev compute done with buf^1
    if (k0 + 64 < K) stage(k0 + 64, buf ^ 1);   // overlap next DMA with WMMAs

    unsigned short* As = smem + buf * 9216;
    unsigned short* Bs = smem + 2 * 9216 + buf * 9216;
#pragma unroll
    for (int kk = 0; kk < 64; kk += 32) {
      FragU af[2], bfm[4];
#pragma unroll
      for (int mi = 0; mi < 2; mi++) {
        int base = (wm * 32 + mi * 16 + ln) * 72 + kk + g * 8;
        af[mi].q[0] = *(const uint4*)&As[base];
        af[mi].q[1] = *(const uint4*)&As[base + 16];
      }
#pragma unroll
      for (int ni = 0; ni < 4; ni++) {
        int base = (wn * 64 + ni * 16 + ln) * 72 + kk + g * 8;
        bfm[ni].q[0] = *(const uint4*)&Bs[base];
        bfm[ni].q[1] = *(const uint4*)&Bs[base + 16];
      }
#pragma unroll
      for (int mi = 0; mi < 2; mi++)
#pragma unroll
        for (int ni = 0; ni < 4; ni++)
          acc[mi][ni] = __builtin_amdgcn_wmma_f32_16x16x32_bf16(
              false, af[mi].v, false, bfm[ni].v, (short)0, acc[mi][ni], false, false);
    }
  }

  // C layout: lane (g,ln), VGPR r -> M = 8g+r, N = ln
#pragma unroll
  for (int mi = 0; mi < 2; mi++)
#pragma unroll
    for (int ni = 0; ni < 4; ni++)
#pragma unroll
      for (int r = 0; r < 8; r++) {
        int row = m0 + wm * 32 + mi * 16 + 8 * g + r;
        int col = n0 + wn * 64 + ni * 16 + ln;
        float vv = acc[mi][ni][r];
        if (OUT_F32) ((float*)Cp)[(size_t)row * N + col] = vv;
        else ((unsigned short*)Cp)[(size_t)row * N + col] = f2bf(vv);
      }
}

// ---------------------------------------------------------------------------
// RoPE (in place, bf16 storage / fp32 math). One thread per (b,s,h) row.
// ---------------------------------------------------------------------------
__global__ void __launch_bounds__(256)
rope_kernel(unsigned short* __restrict__ qr, unsigned short* __restrict__ kr,
            int S, int H, int total)
{
  int row = blockIdx.x * blockDim.x + threadIdx.x;
  if (row >= total) return;
  int s = (row / H) % S;
  unsigned short* q = qr + (size_t)row * 64;
  unsigned short* k = kr + (size_t)row * 64;
  float tpos = (float)s;
#pragma unroll
  for (int i = 0; i < 32; i++) {
    float fr = tpos * __powf(10000.0f, -(float)i * (1.0f / 32.0f));
    float sn, cs;
    __sincosf(fr, &sn, &cs);
    float q1 = bf2f(q[i]), q2 = bf2f(q[i + 32]);
    q[i]      = f2bf(q1 * cs - q2 * sn);
    q[i + 32] = f2bf(q2 * cs + q1 * sn);
    float k1 = bf2f(k[i]), k2 = bf2f(k[i + 32]);
    k[i]      = f2bf(k1 * cs - k2 * sn);
    k[i + 32] = f2bf(k2 * cs + k1 * sn);
  }
}

// ---------------------------------------------------------------------------
// Flash attention, causal. Grid (S/128, H, B). 256 thr = 8 waves; wave owns 16
// query rows (Q 16x192 in registers as 6 A-frags). K tile staged to LDS by
// async DMA (row-major, stride 200); V staged transposed (stride 136) so
// P@V B-frags are contiguous. P bounces through per-wave LDS to flip
// C-layout -> A-layout.
// ---------------------------------------------------------------------------
#define ATT_SCALE 0.07216878364870323f   // 1/sqrt(192)

__global__ void __launch_bounds__(256)
flash_kernel(const unsigned short* __restrict__ qc, const unsigned short* __restrict__ qr,
             const unsigned short* __restrict__ kc, const unsigned short* __restrict__ kr,
             const unsigned short* __restrict__ vv, unsigned short* __restrict__ attn,
             int S)
{
  extern __shared__ unsigned short lds[];
  unsigned short* Kt = lds;                 // [128][200]
  unsigned short* Vt = Kt + 128 * 200;      // [128 d][136 krow]
  unsigned short* Ps = Vt + 128 * 136;      // [8 waves][16][136]

  const int H = 16;
  const int b = blockIdx.z, h = blockIdx.y;
  const int q0 = blockIdx.x * 128;
  const int t = threadIdx.x;
  const int lane = t & 31, wave = t >> 5;
  const int g = lane >> 4, ln = lane & 15;

  // Q fragments: 6 x (16x32); cols 0..127 from q_c, 128..191 from roped q_r
  const int qrow = q0 + wave * 16 + ln;
  const size_t qcb = ((size_t)(b * S + qrow) * H + h) * 128;
  const size_t qrb = ((size_t)(b * S + qrow) * H + h) * 64;
  FragU qf[6];
#pragma unroll
  for (int f = 0; f < 6; f++) {
    const unsigned short* src = (f < 4) ? (qc + qcb + f * 32) : (qr + qrb + (f - 4) * 32);
    qf[f].q[0] = *(const uint4*)(src + g * 8);
    qf[f].q[1] = *(const uint4*)(src + 16 + g * 8);
  }

  const v8f vzero = {};
  v8f o[8];
#pragma unroll
  for (int i = 0; i < 8; i++) o[i] = vzero;
  float mrow[8], lrow[8];
#pragma unroll
  for (int r = 0; r < 8; r++) { mrow[r] = -1e30f; lrow[r] = 0.0f; }

  const int ntiles = q0 / 128 + 1;
  for (int kt = 0; kt < ntiles; kt++) {
    const int kt0 = kt * 128;
    __syncthreads();   // previous tile fully consumed before overwrite

    // ---- K tile: k_c (128x128) + roped k_r (128x64) ----
#if USE_ASYNC_LDS
#pragma unroll
    for (int i = 0; i < 8; i++) {
      int c = i * 256 + t;
      int row = c >> 4, c8 = (c & 15) * 8;
      ASYNC_CP_B128(kc + ((size_t)(b * S + kt0 + row) * H + h) * 128 + c8,
                    &Kt[row * 200 + c8]);
    }
#pragma unroll
    for (int i = 0; i < 4; i++) {
      int c = i * 256 + t;
      int row = c >> 3, c8 = (c & 7) * 8;
      ASYNC_CP_B128(kr + ((size_t)(b * S + kt0 + row) * H + h) * 64 + c8,
                    &Kt[row * 200 + 128 + c8]);
    }
#else
#pragma unroll
    for (int i = 0; i < 8; i++) {
      int c = i * 256 + t;
      int row = c >> 4, c8 = (c & 15) * 8;
      uint4 d = *(const uint4*)(kc + ((size_t)(b * S + kt0 + row) * H + h) * 128 + c8);
      *(uint4*)&Kt[row * 200 + c8] = d;
    }
#pragma unroll
    for (int i = 0; i < 4; i++) {
      int c = i * 256 + t;
      int row = c >> 3, c8 = (c & 7) * 8;
      uint4 d = *(const uint4*)(kr + ((size_t)(b * S + kt0 + row) * H + h) * 64 + c8);
      *(uint4*)&Kt[row * 200 + 128 + c8] = d;
    }
#endif
    // ---- V tile, transposed through VGPRs (overlaps with async K DMA) ----
#pragma unroll
    for (int i = 0; i < 8; i++) {
      int c = i * 256 + t;
      int row = c >> 4, d8 = (c & 15) * 8;
      union { uint4 q; unsigned short u[8]; } u;
      u.q = *(const uint4*)(vv + ((size_t)(b * S + kt0 + row) * H + h) * 128 + d8);
#pragma unroll
      for (int j = 0; j < 8; j++) Vt[(d8 + j) * 136 + row] = u.u[j];
    }
    wait_async0();
    __syncthreads();

    // ---- scores S = Q K^T : 8 col-tiles x 6 k-steps ----
    v8f sa[8];
#pragma unroll
    for (int i = 0; i < 8; i++) sa[i] = vzero;
#pragma unroll
    for (int tt = 0; tt < 8; tt++)
#pragma unroll
      for (int f = 0; f < 6; f++) {
        FragU kf_;
        int base = (tt * 16 + ln) * 200 + f * 32 + g * 8;
        kf_.q[0] = *(const uint4*)&Kt[base];
        kf_.q[1] = *(const uint4*)&Kt[base + 16];
        sa[tt] = __builtin_amdgcn_wmma_f32_16x16x32_bf16(
            false, qf[f].v, false, kf_.v, (short)0, sa[tt], false, false);
      }

    // ---- scale + causal mask (only the diagonal tile) ----
    const bool diag = (kt0 == q0);
#pragma unroll
    for (int tt = 0; tt < 8; tt++)
#pragma unroll
      for (int r = 0; r < 8; r++) {
        float val = sa[tt][r] * ATT_SCALE;
        if (diag) {
          int col = kt0 + tt * 16 + ln;
          int rw  = q0 + wave * 16 + 8 * g + r;
          if (col > rw) val = -1e30f;
        }
        sa[tt][r] = val;
      }

    // ---- online softmax (rows live in 16-lane groups; xor<=8 stays inside) ----
    float mnew[8], sc[8];
#pragma unroll
    for (int r = 0; r < 8; r++) {
      float mx = sa[0][r];
#pragma unroll
      for (int tt = 1; tt < 8; tt++) mx = fmaxf(mx, sa[tt][r]);
#pragma unroll
      for (int off = 1; off <= 8; off <<= 1) mx = fmaxf(mx, __shfl_xor(mx, off, 32));
      mnew[r] = fmaxf(mrow[r], mx);
      sc[r] = __expf(mrow[r] - mnew[r]);
    }
    float rsum[8];
#pragma unroll
    for (int r = 0; r < 8; r++) rsum[r] = 0.0f;
#pragma unroll
    for (int tt = 0; tt < 8; tt++)
#pragma unroll
      for (int r = 0; r < 8; r++) {
        float p = __expf(sa[tt][r] - mnew[r]);
        sa[tt][r] = p;
        rsum[r] += p;
      }
#pragma unroll
    for (int r = 0; r < 8; r++) {
#pragma unroll
      for (int off = 1; off <= 8; off <<= 1) rsum[r] += __shfl_xor(rsum[r], off, 32);
      lrow[r] = lrow[r] * sc[r] + rsum[r];
      mrow[r] = mnew[r];
    }
#pragma unroll
    for (int tt = 0; tt < 8; tt++)
#pragma unroll
      for (int r = 0; r < 8; r++) o[tt][r] *= sc[r];

    // ---- P -> per-wave LDS (C-layout write), read back as A-frags ----
    unsigned short* pw = Ps + wave * 16 * 136;
#pragma unroll
    for (int tt = 0; tt < 8; tt++)
#pragma unroll
      for (int r = 0; r < 8; r++)
        pw[(8 * g + r) * 136 + tt * 16 + ln] = f2bf(sa[tt][r]);

#pragma unroll
    for (int kf = 0; kf < 4; kf++) {
      FragU pf;
      int pb = ln * 136 + kf * 32 + g * 8;
      pf.q[0] = *(const uint4*)&pw[pb];
      pf.q[1] = *(const uint4*)&pw[pb + 16];
#pragma unroll
      for (int tt = 0; tt < 8; tt++) {
        FragU vf;
        int vb = (tt * 16 + ln) * 136 + kf * 32 + g * 8;
        vf.q[0] = *(const uint4*)&Vt[vb];
        vf.q[1] = *(const uint4*)&Vt[vb + 16];
        o[tt] = __builtin_amdgcn_wmma_f32_16x16x32_bf16(
            false, pf.v, false, vf.v, (short)0, o[tt], false, false);
      }
    }
  }

  // ---- normalize and store (b,s,h,d) bf16 ----
#pragma unroll
  for (int r = 0; r < 8; r++) {
    int srow = q0 + wave * 16 + 8 * g + r;
    float inv = 1.0f / lrow[r];
    size_t ob = ((size_t)(b * S + srow) * H + h) * 128;
#pragma unroll
    for (int tt = 0; tt < 8; tt++)
      attn[ob + tt * 16 + ln] = f2bf(o[tt][r] * inv);
  }
}

// ---------------------------------------------------------------------------
extern "C" void kernel_launch(void* const* d_in, const int* in_sizes, int n_in,
                              void* d_out, int out_size, void* d_ws, size_t ws_size,
                              hipStream_t stream)
{
  const float* x     = (const float*)d_in[0];
  const float* w_dq  = (const float*)d_in[1];
  const float* w_uq  = (const float*)d_in[2];
  const float* w_dkv = (const float*)d_in[3];
  const float* w_uk  = (const float*)d_in[4];
  const float* w_uv  = (const float*)d_in[5];
  const float* w_qr  = (const float*)d_in[6];
  const float* w_kr  = (const float*)d_in[7];
  const float* w_out = (const float*)d_in[8];

  const int E = 2048, H = 16, RD = 64, CQ = 1536, CKV = 512;
  const int M = in_sizes[0] / E;   // b*s = 4096
  const int B = 2, S = M / B;

  unsigned short* ws = (unsigned short*)d_ws;
  size_t off = 0;
  auto alloc = [&](size_t n) { unsigned short* p = ws + off; off += n; return p; };
  // bf16 copies of inputs
  unsigned short* xb    = alloc((size_t)M * E);
  unsigned short* bdq   = alloc((size_t)CQ * E);
  unsigned short* buq   = alloc((size_t)E * CQ);
  unsigned short* bdkv  = alloc((size_t)CKV * E);
  unsigned short* buk   = alloc((size_t)E * CKV);
  unsigned short* buv   = alloc((size_t)E * CKV);
  unsigned short* bqr   = alloc((size_t)(H * RD) * CQ);
  unsigned short* bkr   = alloc((size_t)(H * RD) * E);
  unsigned short* bout  = alloc((size_t)E * E);
  // intermediates
  unsigned short* c_q  = alloc((size_t)M * CQ);
  unsigned short* c_kv = alloc((size_t)M * CKV);
  unsigned short* q_c  = alloc((size_t)M * E);
  unsigned short* k_c  = alloc((size_t)M * E);
  unsigned short* vbuf = alloc((size_t)M * E);
  unsigned short* q_r  = alloc((size_t)M * H * RD);
  unsigned short* k_r  = alloc((size_t)M * H * RD);
  unsigned short* attn = alloc((size_t)M * E);
  (void)ws_size; (void)n_in; (void)out_size;

  // one-time fp32 -> bf16 conversions
  auto cvt = [&](const float* s, unsigned short* d, size_t n) {
    int n8 = (int)(n / 8);
    cvt_f32_bf16<<<(n8 + 255) / 256, 256, 0, stream>>>(s, d, n8);
  };
  cvt(x,     xb,   (size_t)M * E);
  cvt(w_dq,  bdq,  (size_t)CQ * E);
  cvt(w_uq,  buq,  (size_t)E * CQ);
  cvt(w_dkv, bdkv, (size_t)CKV * E);
  cvt(w_uk,  buk,  (size_t)E * CKV);
  cvt(w_uv,  buv,  (size_t)E * CKV);
  cvt(w_qr,  bqr,  (size_t)(H * RD) * CQ);
  cvt(w_kr,  bkr,  (size_t)(H * RD) * E);
  cvt(w_out, bout, (size_t)E * E);

  dim3 blk(256);
  const size_t gsh = (size_t)4 * 128 * 72 * sizeof(unsigned short);  // 73728 B
  // down / rope-key projections
  gemm_bf16_wmma<false><<<dim3(CQ / 128, M / 128), blk, gsh, stream>>>(xb, bdq,  c_q,  M, CQ,  E);
  gemm_bf16_wmma<false><<<dim3(CKV / 128, M / 128), blk, gsh, stream>>>(xb, bdkv, c_kv, M, CKV, E);
  gemm_bf16_wmma<false><<<dim3((H * RD) / 128, M / 128), blk, gsh, stream>>>(xb, bkr, k_r, M, H * RD, E);
  // up projections
  gemm_bf16_wmma<false><<<dim3(E / 128, M / 128), blk, gsh, stream>>>(c_q,  buq, q_c,  M, E, CQ);
  gemm_bf16_wmma<false><<<dim3((H * RD) / 128, M / 128), blk, gsh, stream>>>(c_q, bqr, q_r, M, H * RD, CQ);
  gemm_bf16_wmma<false><<<dim3(E / 128, M / 128), blk, gsh, stream>>>(c_kv, buk, k_c,  M, E, CKV);
  gemm_bf16_wmma<false><<<dim3(E / 128, M / 128), blk, gsh, stream>>>(c_kv, buv, vbuf, M, E, CKV);
  // RoPE
  int rows = B * S * H;
  rope_kernel<<<(rows + 255) / 256, 256, 0, stream>>>(q_r, k_r, S, H, rows);
  // causal flash attention
  size_t fsh = (size_t)(128 * 200 + 128 * 136 + 8 * 16 * 136) * sizeof(unsigned short);
  flash_kernel<<<dim3(S / 128, H, B), blk, fsh, stream>>>(q_c, q_r, k_c, k_r, vbuf, attn, S);
  // output projection -> fp32
  gemm_bf16_wmma<true><<<dim3(E / 128, M / 128), blk, gsh, stream>>>(attn, bout, d_out, M, E, E);
}
